// MultiHeadGATLayer_43499428774617
// MI455X (gfx1250) — matrix operand order repaired and checked by reading, not compiled, not run on hardware
//
#include <hip/hip_runtime.h>
#include <hip/hip_bf16.h>
#include <math.h>

#define H_HEADS 4
#define BC      8
#define N_TOK   2048
#define F_IN    128
#define D_HEAD  64
#define E_EDGES 65536
#define NEG_INF_F (-1e16f)
#define SLOPE_F   0.1f

typedef __bf16 bf16_t;
typedef bf16_t v16bf __attribute__((ext_vector_type(16)));
typedef float  v8f   __attribute__((ext_vector_type(8)));

union V16 {
  v16bf    v;
  unsigned u[8];
  __device__ V16() {}
};

__device__ __forceinline__ unsigned bf16bits(float x) {
  unsigned u = __float_as_uint(x);
  return (u + 0x7FFFu + ((u >> 16) & 1u)) >> 16;   // round-to-nearest-even
}
__device__ __forceinline__ unsigned pack2bf(float a, float b) {
  return bf16bits(a) | (bf16bits(b) << 16);
}
__device__ __forceinline__ v8f wmma_bf16(const V16& a, const V16& b, v8f c) {
  return __builtin_amdgcn_wmma_f32_16x16x32_bf16(false, a.v, false, b.v,
                                                 (short)0, c, false, false);
}

// ---------------------------------------------------------------- mask build
__global__ void __launch_bounds__(256) gat_mask_clear(unsigned* __restrict__ m) {
  m[blockIdx.x * 256 + threadIdx.x] = 0u;   // 131072 words = 512 blocks
}

__global__ void __launch_bounds__(256) gat_mask_set(const int* __restrict__ row,
                                                    const int* __restrict__ col,
                                                    unsigned* __restrict__ m, int E) {
  int e = blockIdx.x * 256 + threadIdx.x;
  if (e < E) {
    int r = row[e], c = col[e];
    atomicOr(&m[(size_t)r * (N_TOK / 32) + (c >> 5)], 1u << (c & 31));
  }
}

// ------------------------------------------- W fp32 [H][F][D] -> bf16 WT[H][D][F]
__global__ void __launch_bounds__(256) gat_wprep(const float* __restrict__ Wsrc,
                                                 unsigned short* __restrict__ WT) {
  const int idx = blockIdx.x * 256 + threadIdx.x;  // 32768 elements
  const int hh = idx >> 13;
  const int r  = idx & 8191;
  const int d  = r >> 7;
  const int f  = r & 127;
  WT[idx] = (unsigned short)bf16bits(Wsrc[(size_t)(hh * F_IN + f) * D_HEAD + d]);
}

// ------------------------------------------- h fp32 -> bf16, same layout (8 MB -> 4 MB)
__global__ void __launch_bounds__(256) gat_hprep(const float* __restrict__ hsrc,
                                                 unsigned short* __restrict__ hbf) {
  const size_t base = ((size_t)blockIdx.x * 256 + threadIdx.x) * 8;  // 2M floats / 8
  float4 a = *(const float4*)(hsrc + base);
  float4 b = *(const float4*)(hsrc + base + 4);
  uint4 o;
  o.x = pack2bf(a.x, a.y);
  o.y = pack2bf(a.z, a.w);
  o.z = pack2bf(b.x, b.y);
  o.w = pack2bf(b.z, b.w);
  *(uint4*)(hbf + base) = o;
}

// ------------------------------------------------------- projection: z = h@W
// one wave per 16-row tile of one (head, batch); z stored bf16 [H*BC][N][64]
__global__ void __launch_bounds__(256) gat_proj(const unsigned short* __restrict__ hbf,
                                                const unsigned short* __restrict__ WT,
                                                unsigned short* __restrict__ z) {
  const int wave = threadIdx.x >> 5;
  const int lane = threadIdx.x & 31;
  const int tile = blockIdx.x * 8 + wave;       // 4096 tiles total
  const int hh   = tile >> 10;                  // / (BC * 128)
  const int rem  = tile & 1023;
  const int b    = rem >> 7;
  const int rowbase = (rem & 127) << 4;
  const int ml = lane & 15;
  const int kh = lane >> 4;

  // A = 16 rows of h (bf16), WMMA-A layout = contiguous f segments
  V16 A[4];
  const unsigned short* hrow = hbf + (size_t)(b * N_TOK + rowbase + ml) * F_IN;
#pragma unroll
  for (int c = 0; c < 4; ++c) {
    *(uint4*)(&A[c].u[0]) = *(const uint4*)(hrow + c * 32 + kh * 16);
    *(uint4*)(&A[c].u[4]) = *(const uint4*)(hrow + c * 32 + kh * 16 + 8);
  }

  const unsigned short* WTh = WT + (size_t)hh * D_HEAD * F_IN;
  unsigned short* zhb = z + (size_t)(hh * BC + b) * N_TOK * D_HEAD;

#pragma unroll
  for (int s = 0; s < 4; ++s) {                 // D subtiles of 16
    v8f acc = {0.f,0.f,0.f,0.f,0.f,0.f,0.f,0.f};
#pragma unroll
    for (int c = 0; c < 4; ++c) {
      // B-layout = 16 consecutive f at fixed d: contiguous in WT
      V16 B;
      const unsigned short* wp = WTh + (size_t)(s * 16 + ml) * F_IN + c * 32 + kh * 16;
      *(uint4*)(&B.u[0]) = *(const uint4*)(wp);
      *(uint4*)(&B.u[4]) = *(const uint4*)(wp + 8);
      acc = wmma_bf16(A[c], B, acc);
    }
#pragma unroll
    for (int v = 0; v < 8; ++v) {
      const int r = rowbase + v + 8 * kh;       // C layout: row v + 8*lanehalf
      zhb[(size_t)r * D_HEAD + s * 16 + ml] = (unsigned short)bf16bits(acc[v]);
    }
  }
}

// ---------------------------------------------- fused masked attention (flash)
// block = 8 waves = one (head, batch, 128-row group); 512 blocks
__global__ void __launch_bounds__(256) gat_attn(const unsigned short* __restrict__ z,
                                                const unsigned* __restrict__ mask,
                                                float* __restrict__ out) {
  __shared__ __align__(16) unsigned short kv_row[128 * 64];   // K/V row-major
  __shared__ __align__(16) unsigned short kv_colT[64 * 128];  // K/V transposed
  __shared__ __align__(16) unsigned short p_buf[8][16 * 32];  // per-wave P relayout

  const int tid  = threadIdx.x;
  const int wave = tid >> 5;
  const int lane = tid & 31;
  const int ml   = lane & 15;
  const int kh   = lane >> 4;

  const int hb = blockIdx.x >> 4;               // 0..31 = (head, batch)
  const int rg = blockIdx.x & 15;
  const int hh = hb >> 3;
  const int b  = hb & 7;
  const int rowbase = rg * 128 + wave * 16;

  const unsigned short* zhb = z + (size_t)hb * N_TOK * D_HEAD;

  // Q tile (16 x 64) in WMMA-A layout, 2 k-chunks of 32
  V16 Q[2];
  {
    const unsigned short* qrow = zhb + (size_t)(rowbase + ml) * D_HEAD;
#pragma unroll
    for (int c = 0; c < 2; ++c) {
      *(uint4*)(&Q[c].u[0]) = *(const uint4*)(qrow + c * 32 + kh * 16);
      *(uint4*)(&Q[c].u[4]) = *(const uint4*)(qrow + c * 32 + kh * 16 + 8);
    }
  }

  const v8f zero8 = {0.f,0.f,0.f,0.f,0.f,0.f,0.f,0.f};
  v8f acc[4];
#pragma unroll
  for (int d = 0; d < 4; ++d) acc[d] = zero8;
  float mrow[8], lrow[8];
#pragma unroll
  for (int v = 0; v < 8; ++v) { mrow[v] = -__builtin_inff(); lrow[v] = 0.f; }

  for (int chunk = 0; chunk < 16; ++chunk) {
    const int colbase = chunk * 128;
    __syncthreads();
    {   // cooperative load of 128x64 bf16 K/V block: row-major + transposed.
        // Row pairs -> transposed writes become packed u32 stores.
      const uint4* gz = (const uint4*)(zhb + (size_t)colbase * D_HEAD);
      unsigned* colT32 = (unsigned*)kv_colT;
#pragma unroll
      for (int i = 0; i < 2; ++i) {
        const int idx = tid + 256 * i;          // 512 (row-pair, 16B-unit) jobs
        const int rp  = idx >> 3;               // row pair 0..63
        const int cu  = idx & 7;                // 16-byte unit within row
        const int r0  = rp * 2;
        uint4 va = gz[r0 * 8 + cu];
        uint4 vb = gz[(r0 + 1) * 8 + cu];
        *(uint4*)(&kv_row[(r0 * 8 + cu) * 8])       = va;
        *(uint4*)(&kv_row[((r0 + 1) * 8 + cu) * 8]) = vb;
        const int cb = cu * 8;
        unsigned wa[4] = {va.x, va.y, va.z, va.w};
        unsigned wb[4] = {vb.x, vb.y, vb.z, vb.w};
#pragma unroll
        for (int j = 0; j < 4; ++j) {
          colT32[(cb + 2 * j) * 64 + rp]     = (wa[j] & 0xFFFFu) | (wb[j] << 16);
          colT32[(cb + 2 * j + 1) * 64 + rp] = (wa[j] >> 16) | (wb[j] & 0xFFFF0000u);
        }
      }
    }
    __syncthreads();

    for (int sub = 0; sub < 4; ++sub) {         // 32 columns at a time
      // ---- S = Q . K^T  (two 16x16 tiles, k = 64 in 2 steps)
      v8f s0 = zero8, s1 = zero8;
#pragma unroll
      for (int c = 0; c < 2; ++c) {
        V16 B0, B1;
        const unsigned short* k0p = &kv_row[(sub * 32 + ml) * 64 + c * 32 + kh * 16];
        const unsigned short* k1p = &kv_row[(sub * 32 + 16 + ml) * 64 + c * 32 + kh * 16];
        *(uint4*)(&B0.u[0]) = *(const uint4*)(k0p);
        *(uint4*)(&B0.u[4]) = *(const uint4*)(k0p + 8);
        *(uint4*)(&B1.u[0]) = *(const uint4*)(k1p);
        *(uint4*)(&B1.u[4]) = *(const uint4*)(k1p + 8);
        s0 = wmma_bf16(Q[c], B0, s0);
        s1 = wmma_bf16(Q[c], B1, s1);
      }

      // ---- LeakyReLU + adjacency mask + online softmax
      const int wc = (colbase + sub * 32) >> 5; // one mask word covers both tiles
      float alpha[8];
#pragma unroll
      for (int v = 0; v < 8; ++v) {
        const int r = rowbase + v + 8 * kh;
        const unsigned mw = mask[(size_t)r * (N_TOK / 32) + wc];
        float x0 = s0[v]; x0 = (x0 > 0.f) ? x0 : SLOPE_F * x0;
        float x1 = s1[v]; x1 = (x1 > 0.f) ? x1 : SLOPE_F * x1;
        const bool e0 = (((mw >> ml) & 1u) != 0u) && (x0 != 0.f);
        const bool e1 = (((mw >> (16 + ml)) & 1u) != 0u) && (x1 != 0.f);
        x0 = e0 ? x0 : NEG_INF_F;
        x1 = e1 ? x1 : NEG_INF_F;
        float mx = fmaxf(x0, x1);               // row lives in one 16-lane half
#pragma unroll
        for (int off = 1; off < 16; off <<= 1) mx = fmaxf(mx, __shfl_xor(mx, off, 32));
        const float mnew = fmaxf(mrow[v], mx);
        const float a  = __expf(mrow[v] - mnew);
        const float p0 = __expf(x0 - mnew);
        const float p1 = __expf(x1 - mnew);
        float ps = p0 + p1;
#pragma unroll
        for (int off = 1; off < 16; off <<= 1) ps += __shfl_xor(ps, off, 32);
        lrow[v] = lrow[v] * a + ps;
        mrow[v] = mnew;
        alpha[v] = a;
        p_buf[wave][(v + 8 * kh) * 32 + ml]      = (unsigned short)bf16bits(p0);
        p_buf[wave][(v + 8 * kh) * 32 + 16 + ml] = (unsigned short)bf16bits(p1);
      }
#pragma unroll
      for (int d = 0; d < 4; ++d)
#pragma unroll
        for (int v = 0; v < 8; ++v) acc[d][v] *= alpha[v];

      __builtin_amdgcn_wave_barrier();          // order P store -> P reload

      // ---- O += P . V   (P relayout C->A via wave-private LDS)
      V16 P;
      const unsigned short* pm = &p_buf[wave][ml * 32];
      *(uint4*)(&P.u[0]) = *(const uint4*)(pm + kh * 16);
      *(uint4*)(&P.u[4]) = *(const uint4*)(pm + kh * 16 + 8);
#pragma unroll
      for (int d = 0; d < 4; ++d) {
        V16 Vb;
        const unsigned short* vp = &kv_colT[(d * 16 + ml) * 128 + sub * 32 + kh * 16];
        *(uint4*)(&Vb.u[0]) = *(const uint4*)(vp);
        *(uint4*)(&Vb.u[4]) = *(const uint4*)(vp + 8);
        acc[d] = wmma_bf16(P, Vb, acc[d]);
      }
    }
  }

  // out[b][n][hh*64 + d], fp32
#pragma unroll
  for (int d = 0; d < 4; ++d)
#pragma unroll
    for (int v = 0; v < 8; ++v) {
      const int r = rowbase + v + 8 * kh;
      out[(size_t)(b * N_TOK + r) * (H_HEADS * D_HEAD) + hh * D_HEAD + d * 16 + ml]
          = acc[d][v] / lrow[v];
    }
}

// ---------------------------------------------------------------------------
extern "C" void kernel_launch(void* const* d_in, const int* in_sizes, int n_in,
                              void* d_out, int out_size, void* d_ws, size_t ws_size,
                              hipStream_t stream) {
  const float* h   = (const float*)d_in[0];
  const float* W   = (const float*)d_in[1];
  const int*   row = (const int*)d_in[2];
  const int*   col = (const int*)d_in[3];
  float* out = (float*)d_out;

  // workspace: z (bf16, 8 MB) | mask bitmap (512 KB) | WT (64 KB) | hbf (4 MB)
  char* ws = (char*)d_ws;
  unsigned short* z  = (unsigned short*)ws;
  unsigned* mask     = (unsigned*)(ws + (size_t)H_HEADS * BC * N_TOK * D_HEAD * 2);
  unsigned short* WT = (unsigned short*)((char*)mask + (size_t)N_TOK * N_TOK / 8);
  unsigned short* hbf = (unsigned short*)((char*)WT + (size_t)H_HEADS * F_IN * D_HEAD * 2);

  gat_wprep<<<(H_HEADS * F_IN * D_HEAD) / 256, 256, 0, stream>>>(W, WT);
  gat_hprep<<<(BC * N_TOK * F_IN) / (256 * 8), 256, 0, stream>>>(h, hbf);
  gat_mask_clear<<<(N_TOK * N_TOK / 32) / 256, 256, 0, stream>>>(mask);
  gat_mask_set<<<(E_EDGES + 255) / 256, 256, 0, stream>>>(row, col, mask, E_EDGES);
  gat_proj<<<512, 256, 0, stream>>>(hbf, WT, z);
  gat_attn<<<512, 256, 0, stream>>>(z, mask, out);
}
